// fftatt_31044023615909
// MI455X (gfx1250) — compile-verified, matrix operand order checked by reading, not run on hardware
//
#include <hip/hip_runtime.h>

// =====================================================================
// fftatt for MI455X (gfx1250, wave32, WMMA).
// Pipeline (all on `stream`, deterministic, no atomics):
//  0. prep: f32 weights -> bf16 (wq/wk/wv/wd row-major; wc/wi -> [oc][tap][ic]);
//     zero-bias vector for the attention out-GEMM.
//  1. yd   = conv1x1(y, wd)                      [WMMA bf16]
//  2. row FFT(yd) -> tmpc      (w fftshift folded into store)
//  3. col FFT     -> dft_pre   (b/c/h fftshift folded into store, 1/256 ortho)
//  4. dft  = conv3x3(dft_pre, wc)                [implicit-GEMM WMMA bf16]
//  5. xq = conv1x1(x,wq); yk = conv1x1(dft[:32],wk); yv = conv1x1(dft[32:],wv)
//  6. att partials (split-K=128 deterministic) -> reduce + softmax(axis=c) -> bf16
//  7. attout = att x yv                          [WMMA bf16]
//  8. wiout = conv3x3(relu(dft), wi)             [WMMA bf16]
//  9. row IFFT (ifftshift folded into loads) -> tmpc ; col IFFT -> abs -> out += attout
// =====================================================================

#define DEVFN __device__ __forceinline__

typedef __attribute__((ext_vector_type(16))) __bf16 v16bf;
typedef __attribute__((ext_vector_type(2)))  __bf16 v2bf;
typedef __attribute__((ext_vector_type(8)))  float  v8f;

union Frag16 { v16bf v; unsigned u[8]; };

static constexpr int  CH   = 32;
static constexpr int  HH   = 256;
static constexpr int  WW   = 256;
static constexpr int  HWC  = HH * WW;      // 65536
static constexpr int  NB   = 4;

// ---- f32 -> bf16 round-to-nearest-even, packed pair helpers ----
DEVFN unsigned bfr(float x) {
  unsigned u = __builtin_bit_cast(unsigned, x);
  return (u + 0x7FFFu + ((u >> 16) & 1u)) >> 16;
}
DEVFN unsigned pack2_bf16(float lo, float hi) {
#if defined(__has_builtin)
#if __has_builtin(__builtin_amdgcn_cvt_pk_bf16_f32)
  v2bf p = __builtin_amdgcn_cvt_pk_bf16_f32(lo, hi);
  return __builtin_bit_cast(unsigned, p);
#else
  return bfr(lo) | (bfr(hi) << 16);
#endif
#else
  return bfr(lo) | (bfr(hi) << 16);
#endif
}

DEVFN v8f wmma_bf16(const Frag16& A, const Frag16& B, v8f C) {
  return __builtin_amdgcn_wmma_f32_16x16x32_bf16(
      /*neg_a=*/false, A.v, /*neg_b=*/false, B.v,
      /*c_mod=*/(short)0, C, /*reuse_a=*/false, /*reuse_b=*/false);
}

// =====================================================================
// 0) weight prep: f32 -> bf16 layouts + zero bias vector
//    w1x1b: [wq|wk|wv|wd], each 32x32 row-major (o major, c fastest)
//    wcb/wib: [oc][tap][ic] bf16 (ic contiguous -> aligned uint4 A-frag loads)
// =====================================================================
__global__ __launch_bounds__(256)
void k_prep(const float* __restrict__ wq, const float* __restrict__ wk,
            const float* __restrict__ wv, const float* __restrict__ wd,
            const float* __restrict__ wc, const float* __restrict__ wi,
            unsigned short* __restrict__ w1x1b,
            unsigned short* __restrict__ wcb, unsigned short* __restrict__ wib,
            float* __restrict__ zbias) {
  int stride = gridDim.x * blockDim.x;
  int tid0 = blockIdx.x * blockDim.x + threadIdx.x;
  if (tid0 < 32) zbias[tid0] = 0.0f;
  for (int i = tid0; i < 4 * 1024; i += stride) {
    int wsel = i >> 10, j = i & 1023;
    const float* s = (wsel == 0) ? wq : (wsel == 1) ? wk : (wsel == 2) ? wv : wd;
    w1x1b[i] = (unsigned short)bfr(s[j]);
  }
  for (int i = tid0; i < 2 * 36864; i += stride) {
    int wsel = i / 36864, j = i % 36864;
    int ic = j & 63, t2 = j >> 6;
    int tap = t2 % 9, oc = t2 / 9;
    const float* s = wsel ? wi : wc;
    unsigned short* d = wsel ? wib : wcb;
    d[j] = (unsigned short)bfr(s[(oc * 64 + ic) * 9 + tap]);
  }
}

// =====================================================================
// conv1x1 / att-out GEMM:  out[b,o,n] = sum_c A[o,c] * in[b,cin0+c,n] + bias[o]
// One wave = one 16-wide n tile, both 16-row M tiles, single K=32 WMMA each.
// A is bf16 (weights or per-batch softmaxed attention, a_bstride elements).
// bias must be a valid 32-float pointer (zero vector for the att GEMM).
// grid: B*HW/(16*8) blocks x 256 threads.
// =====================================================================
__global__ __launch_bounds__(256)
void k_conv1x1(const float* __restrict__ in, long in_bstride, int in_coff,
               const unsigned short* __restrict__ Abf, long a_bstride,
               const float* __restrict__ bias, float* __restrict__ out) {
  int wave = threadIdx.x >> 5, lane = threadIdx.x & 31;
  int nl = lane & 15, half = lane >> 4;
  int nglob = blockIdx.x * 128 + wave * 16;
  int b = nglob >> 16, n0 = nglob & (HWC - 1);
  const float* inb = in + (long)b * in_bstride + (long)in_coff * HWC + n0 + nl;
  __builtin_prefetch(inb + 16, 0, 0);

  // B fragment: lane column n, K = channels 16*half .. 16*half+15
  Frag16 Bf;
  int kc0 = 16 * half;
#pragma unroll
  for (int v = 0; v < 8; ++v) {
    float lo = inb[(long)(kc0 + 2 * v) * HWC];
    float hi = inb[(long)(kc0 + 2 * v + 1) * HWC];
    Bf.u[v] = pack2_bf16(lo, hi);
  }

  const unsigned short* Ab = Abf + (long)b * a_bstride;
  float* outb = out + (long)b * (32L * HWC);
#pragma unroll
  for (int mt = 0; mt < 2; ++mt) {
    Frag16 Af;
    const uint4* ap = (const uint4*)(Ab + ((mt * 16 + nl) * 32 + kc0));
    uint4 a0 = ap[0], a1 = ap[1];
    Af.u[0] = a0.x; Af.u[1] = a0.y; Af.u[2] = a0.z; Af.u[3] = a0.w;
    Af.u[4] = a1.x; Af.u[5] = a1.y; Af.u[6] = a1.z; Af.u[7] = a1.w;
    v8f acc = {};
    acc = wmma_bf16(Af, Bf, acc);

    // bias for rows mt*16 + 8*half + (0..7): two aligned float4 loads
    float bv[8];
    const float4* bp = (const float4*)(bias + mt * 16 + 8 * half);
    *(float4*)(bv) = bp[0];
    *(float4*)(bv + 4) = bp[1];
#pragma unroll
    for (int r = 0; r < 8; ++r) {
      int oc = mt * 16 + r + 8 * half;
      outb[(long)oc * HWC + n0 + nl] = acc[r] + bv[r];
    }
  }
}

// =====================================================================
// conv3x3 64->64 SAME, implicit GEMM (M=64 oc, N=64 px row segment, K=576).
// Block: (b, h, 64-wide w segment). LDS: 3 rows x 66 px x 64 ic bf16 packed
// as u32 pairs, px stride 33 dwords (odd -> bank-conflict-free).
// 8 waves: wave -> (mtile = w>>1, ntiles {2*(w&1), 2*(w&1)+1}); 36 WMMAs/wave.
// =====================================================================
__global__ __launch_bounds__(256)
void k_conv3x3(const float* __restrict__ in, const unsigned short* __restrict__ Wbf,
               const float* __restrict__ bias, float* __restrict__ out, int relu_in) {
  __shared__ unsigned lds3[3 * 66 * 33];
  int bw = blockIdx.x & 3, h = blockIdx.x >> 2, b = blockIdx.y;
  int w0 = bw * 64;
  const float* inb = in + (long)b * 64 * HWC;

  for (int i = threadIdx.x; i < 3 * 66 * 32; i += 256) {
    int px = i % 66, icp = (i / 66) & 31, r = i / (66 * 32);
    int hh = h - 1 + r, ww = w0 - 1 + px;
    float lo = 0.f, hi = 0.f;
    if ((unsigned)hh < 256u && (unsigned)ww < 256u) {
      long base = (long)hh * WW + ww;
      lo = inb[(long)(2 * icp) * HWC + base];
      hi = inb[(long)(2 * icp + 1) * HWC + base];
      if (relu_in) { lo = fmaxf(lo, 0.f); hi = fmaxf(hi, 0.f); }
    }
    lds3[(r * 66 + px) * 33 + icp] = pack2_bf16(lo, hi);
  }
  __syncthreads();

  int wave = threadIdx.x >> 5, lane = threadIdx.x & 31;
  int nl = lane & 15, half = lane >> 4;
  int mt = wave >> 1;             // 0..3 -> oc tile
  int ntb = (wave & 1) * 2;       // px tiles ntb, ntb+1
  v8f acc0 = {}, acc1 = {};

#pragma unroll
  for (int tap = 0; tap < 9; ++tap) {
    int rr = tap / 3, dw = tap % 3;
#pragma unroll
    for (int kc = 0; kc < 2; ++kc) {
      Frag16 Af;
      const uint4* ap =
          (const uint4*)(Wbf + ((mt * 16 + nl) * 9 + tap) * 64 + kc * 32 + half * 16);
      uint4 a0 = ap[0], a1 = ap[1];
      Af.u[0] = a0.x; Af.u[1] = a0.y; Af.u[2] = a0.z; Af.u[3] = a0.w;
      Af.u[4] = a1.x; Af.u[5] = a1.y; Af.u[6] = a1.z; Af.u[7] = a1.w;

      int kb = kc * 16 + half * 8;
      int px0 = ntb * 16 + nl + dw;
      Frag16 Bf0, Bf1;
#pragma unroll
      for (int v = 0; v < 8; ++v) {
        Bf0.u[v] = lds3[(rr * 66 + px0) * 33 + kb + v];
        Bf1.u[v] = lds3[(rr * 66 + px0 + 16) * 33 + kb + v];
      }
      acc0 = wmma_bf16(Af, Bf0, acc0);
      acc1 = wmma_bf16(Af, Bf1, acc1);
    }
  }

  float bv[8];
  const float4* bp = (const float4*)(bias + mt * 16 + 8 * half);
  *(float4*)(bv) = bp[0];
  *(float4*)(bv + 4) = bp[1];
  float* outb = out + (long)b * 64 * HWC + (long)h * WW + w0;
#pragma unroll
  for (int r = 0; r < 8; ++r) {
    int oc = mt * 16 + r + 8 * half;
    outb[(long)oc * HWC + ntb * 16 + nl] = acc0[r] + bv[r];
    outb[(long)oc * HWC + (ntb + 1) * 16 + nl] = acc1[r] + bv[r];
  }
}

// =====================================================================
// radix-2 256-pt FFT stages in LDS (128 threads per row, bit-reversed input).
// sign = -1 forward, +1 inverse.
// =====================================================================
DEVFN void fft256_stages(float2* a, int t, float sign) {
  for (int s = 1; s <= 8; ++s) {
    int hm = 1 << (s - 1);
    int pos = t & (hm - 1);
    int i0 = ((t >> (s - 1)) << s) + pos;
    int i1 = i0 + hm;
    float ang = sign * (6.283185307179586f * (float)pos / (float)(2 * hm));
    float sw, cw;
    __sincosf(ang, &sw, &cw);
    float2 x0 = a[i0], x1 = a[i1];
    float tr = x1.x * cw - x1.y * sw;
    float ti = x1.x * sw + x1.y * cw;
    a[i0] = make_float2(x0.x + tr, x0.y + ti);
    a[i1] = make_float2(x0.x - tr, x0.y - ti);
    __syncthreads();
  }
}

// ---- forward row FFT (real input), fftshift(w) folded into the store ----
__global__ __launch_bounds__(256)
void k_fft_rows(const float* __restrict__ yd, float2* __restrict__ tmpc) {
  __shared__ float2 row[2][256];
  int rl = threadIdx.x >> 7, t = threadIdx.x & 127;
  long gr = (long)blockIdx.x * 2 + rl;     // (b*32+c)*256 + h
  const float* src = yd + gr * WW;
#pragma unroll
  for (int q = 0; q < 2; ++q) {
    int e = t + q * 128;
    int rv = __brev((unsigned)e) >> 24;
    row[rl][rv] = make_float2(src[e], 0.f);
  }
  __syncthreads();
  fft256_stages(row[rl], t, -1.0f);
#pragma unroll
  for (int q = 0; q < 2; ++q) {
    int e = t + q * 128;
    float2 v = row[rl][e];
    tmpc[gr * WW + ((e + 128) & 255)] = make_float2(v.x * 0.0625f, v.y * 0.0625f);
  }
}

// ---- forward col FFT, fftshift(b,c,h) folded into the store ----
__global__ __launch_bounds__(256)
void k_fft_cols(const float2* __restrict__ tmpc, float* __restrict__ dft_pre) {
  __shared__ float2 col[256][17];
  int wg = blockIdx.x & 15;
  int c = (blockIdx.x >> 4) & 31;
  int b = blockIdx.x >> 9;
  const float2* src = tmpc + ((long)(b * 32 + c)) * HWC + wg * 16;
  for (int i = threadIdx.x; i < 4096; i += 256) {
    int cl = i & 15, hh = i >> 4;
    int rv = __brev((unsigned)hh) >> 24;
    col[rv][cl] = src[(long)hh * WW + cl];
  }
  __syncthreads();
  int cl = threadIdx.x & 15, tt = threadIdx.x >> 4;
  for (int s = 1; s <= 8; ++s) {
    int hm = 1 << (s - 1);
#pragma unroll
    for (int q = 0; q < 8; ++q) {
      int tb = tt + q * 16;
      int pos = tb & (hm - 1);
      int i0 = ((tb >> (s - 1)) << s) + pos;
      int i1 = i0 + hm;
      float ang = -6.283185307179586f * (float)pos / (float)(2 * hm);
      float sw, cw;
      __sincosf(ang, &sw, &cw);
      float2 x0 = col[i0][cl], x1 = col[i1][cl];
      float tr = x1.x * cw - x1.y * sw;
      float ti = x1.x * sw + x1.y * cw;
      col[i0][cl] = make_float2(x0.x + tr, x0.y + ti);
      col[i1][cl] = make_float2(x0.x - tr, x0.y - ti);
    }
    __syncthreads();
  }
  int bsh = (b + 2) & 3, csh = (c + 16) & 31;
  float* dre = dft_pre + ((long)bsh * 64 + csh) * HWC;
  float* dim = dre + 32L * HWC;
  for (int i = threadIdx.x; i < 4096; i += 256) {
    int cl2 = i & 15, hh = i >> 4;
    float2 v = col[hh][cl2];
    long o = (long)((hh + 128) & 255) * WW + wg * 16 + cl2;
    dre[o] = v.x * 0.0625f;
    dim[o] = v.y * 0.0625f;
  }
}

// ---- inverse row FFT; ifftshift(b,ch,h,w) over [B,64,H,W] folded into loads.
//      shifted re-plane(c) = in[(b+2)%4][c+32], shifted im-plane(c) = in[..][c].
__global__ __launch_bounds__(256)
void k_ifft_rows(const float* __restrict__ wiout, float2* __restrict__ tmpc) {
  __shared__ float2 row[2][256];
  int rl = threadIdx.x >> 7, t = threadIdx.x & 127;
  long idx = (long)blockIdx.x * 2 + rl;     // (b*32+c)*256 + h
  int h = (int)(idx & 255);
  int c = (int)((idx >> 8) & 31);
  int b = (int)(idx >> 13);
  int bsh = (b + 2) & 3, hh = (h + 128) & 255;
  const float* pre = wiout + (((long)bsh * 64 + (c + 32)) * HH + hh) * (long)WW;
  const float* pim = wiout + (((long)bsh * 64 + c) * HH + hh) * (long)WW;
#pragma unroll
  for (int q = 0; q < 2; ++q) {
    int e = t + q * 128;
    int we = (e + 128) & 255;
    int rv = __brev((unsigned)e) >> 24;
    row[rl][rv] = make_float2(pre[we], pim[we]);
  }
  __syncthreads();
  fft256_stages(row[rl], t, +1.0f);
#pragma unroll
  for (int q = 0; q < 2; ++q) {
    int e = t + q * 128;
    float2 v = row[rl][e];
    tmpc[idx * WW + e] = make_float2(v.x * 0.0625f, v.y * 0.0625f);
  }
}

// ---- inverse col FFT + abs + add attention output -> final out ----
__global__ __launch_bounds__(256)
void k_ifft_cols(const float2* __restrict__ tmpc, const float* __restrict__ attout,
                 float* __restrict__ out) {
  __shared__ float2 col[256][17];
  int wg = blockIdx.x & 15;
  int c = (blockIdx.x >> 4) & 31;
  int b = blockIdx.x >> 9;
  const float2* src = tmpc + ((long)(b * 32 + c)) * HWC + wg * 16;
  for (int i = threadIdx.x; i < 4096; i += 256) {
    int cl = i & 15, hh = i >> 4;
    int rv = __brev((unsigned)hh) >> 24;
    col[rv][cl] = src[(long)hh * WW + cl];
  }
  __syncthreads();
  int cl = threadIdx.x & 15, tt = threadIdx.x >> 4;
  for (int s = 1; s <= 8; ++s) {
    int hm = 1 << (s - 1);
#pragma unroll
    for (int q = 0; q < 8; ++q) {
      int tb = tt + q * 16;
      int pos = tb & (hm - 1);
      int i0 = ((tb >> (s - 1)) << s) + pos;
      int i1 = i0 + hm;
      float ang = 6.283185307179586f * (float)pos / (float)(2 * hm);
      float sw, cw;
      __sincosf(ang, &sw, &cw);
      float2 x0 = col[i0][cl], x1 = col[i1][cl];
      float tr = x1.x * cw - x1.y * sw;
      float ti = x1.x * sw + x1.y * cw;
      col[i0][cl] = make_float2(x0.x + tr, x0.y + ti);
      col[i1][cl] = make_float2(x0.x - tr, x0.y - ti);
    }
    __syncthreads();
  }
  for (int i = threadIdx.x; i < 4096; i += 256) {
    int cl2 = i & 15, hh = i >> 4;
    float2 v = col[hh][cl2];
    float re = v.x * 0.0625f, im = v.y * 0.0625f;
    long o = (((long)b * 32 + c) * HH + hh) * (long)WW + wg * 16 + cl2;
    out[o] = attout[o] + sqrtf(re * re + im * im);
  }
}

// =====================================================================
// attention: att[b,c,d] = sum_n xq[b,c,n]*yk[b,d,n]. Deterministic split-K:
// 128 segments of 512 -> partials[b][seg][32][32], then reduce + softmax(c).
// grid: B*64 blocks, 8 waves: (mt,nt) tile + 2 K sub-segments per block.
// =====================================================================
__global__ __launch_bounds__(256)
void k_att_partial(const float* __restrict__ xq, const float* __restrict__ yk,
                   float* __restrict__ part) {
  int b = blockIdx.x >> 6, ks = blockIdx.x & 63;
  int wave = threadIdx.x >> 5, lane = threadIdx.x & 31;
  int nl = lane & 15, half = lane >> 4;
  int mt = wave & 1, nt = (wave >> 1) & 1, ksl = wave >> 2;
  int seg = ks * 2 + ksl;
  long kbase = (long)seg * 512;
  const float* xa = xq + (long)b * 32 * HWC + (long)(mt * 16 + nl) * HWC + kbase + half * 16;
  const float* yb = yk + (long)b * 32 * HWC + (long)(nt * 16 + nl) * HWC + kbase + half * 16;
  v8f acc = {};
  for (int kk = 0; kk < 16; ++kk) {
    const float* xp = xa + kk * 32;
    const float* yp = yb + kk * 32;
    __builtin_prefetch(xp + 64, 0, 0);
    Frag16 Af, Bf;
#pragma unroll
    for (int v = 0; v < 8; ++v) {
      Af.u[v] = pack2_bf16(xp[2 * v], xp[2 * v + 1]);
      Bf.u[v] = pack2_bf16(yp[2 * v], yp[2 * v + 1]);
    }
    acc = wmma_bf16(Af, Bf, acc);
  }
  float* pp = part + ((long)b * 128 + seg) * 1024;
#pragma unroll
  for (int r = 0; r < 8; ++r) {
    int c = mt * 16 + r + 8 * half;
    pp[c * 32 + nt * 16 + nl] = acc[r];
  }
}

__global__ __launch_bounds__(256)
void k_att_reduce(const float* __restrict__ part, unsigned short* __restrict__ attbf) {
  __shared__ float a[32][33];
  __shared__ float mx[32], sminv[32];
  int b = blockIdx.x;
  const float* pb = part + (long)b * 128 * 1024;
  for (int e = threadIdx.x; e < 1024; e += 256) {
    float s = 0.f;
    for (int sg = 0; sg < 128; ++sg) s += pb[sg * 1024 + e];
    a[e >> 5][e & 31] = s;
  }
  __syncthreads();
  if (threadIdx.x < 32) {
    int d = threadIdx.x;
    float m = -3.4e38f;
    for (int c = 0; c < 32; ++c) m = fmaxf(m, a[c][d]);
    float se = 0.f;
    for (int c = 0; c < 32; ++c) se += __expf(a[c][d] - m);
    mx[d] = m;
    sminv[d] = 1.0f / se;
  }
  __syncthreads();
  for (int e = threadIdx.x; e < 1024; e += 256) {
    int c = e >> 5, d = e & 31;
    float v = __expf(a[c][d] - mx[d]) * sminv[d];
    attbf[(long)b * 1024 + e] = (unsigned short)bfr(v);
  }
}

// =====================================================================
// host launcher
// =====================================================================
extern "C" void kernel_launch(void* const* d_in, const int* in_sizes, int n_in,
                              void* d_out, int out_size, void* d_ws, size_t ws_size,
                              hipStream_t stream) {
  (void)in_sizes; (void)n_in; (void)out_size; (void)ws_size;
  const float* x  = (const float*)d_in[0];
  const float* y  = (const float*)d_in[1];
  const float* wq = (const float*)d_in[2];
  const float* bq = (const float*)d_in[3];
  const float* wk = (const float*)d_in[4];
  const float* bk = (const float*)d_in[5];
  const float* wv = (const float*)d_in[6];
  const float* bv = (const float*)d_in[7];
  const float* wd = (const float*)d_in[8];
  const float* bd = (const float*)d_in[9];
  const float* wc = (const float*)d_in[10];
  const float* bc = (const float*)d_in[11];
  const float* wi = (const float*)d_in[12];
  const float* bi = (const float*)d_in[13];
  float* out = (float*)d_out;

  // workspace carving (stage-DAG aliasing; total ~201 MB)
  char* ws = (char*)d_ws;
  const size_t SZ32 = 33554432ull;   // [B,32,H,W] f32
  const size_t SZ64 = 67108864ull;   // [B,64,H,W] f32 == [B,32,H,W] float2
  size_t o = 0;
  float*  yd_att = (float*)(ws + o); o += SZ32;        // yd, later attout
  float*  tmpdft = (float*)(ws + o); o += SZ64;        // tmpc(fwd) -> dft -> tmpc(inv)
  float*  regB   = (float*)(ws + o); o += SZ64;        // dft_pre -> xq|yk -> wiout
  float*  yv     = (float*)(ws + o); o += SZ32;
  float*  part   = (float*)(ws + o); o += 2097152ull;  // [B,128,32,32] f32
  unsigned short* attbf = (unsigned short*)(ws + o); o += 8192;
  unsigned short* w1x1b = (unsigned short*)(ws + o); o += 8192;
  unsigned short* wcb   = (unsigned short*)(ws + o); o += 73728;
  unsigned short* wib   = (unsigned short*)(ws + o); o += 73728;
  float*  zbias = (float*)(ws + o); o += 256;          // 32 zero floats

  float2* tmpc    = (float2*)tmpdft;
  float*  dft     = tmpdft;
  float*  dft_pre = regB;
  float*  xq      = regB;
  float*  yk      = (float*)((char*)regB + SZ32);
  float*  wiout   = regB;
  float*  attout  = yd_att;

  const int C1BLK = NB * HWC / 128;       // 2048 conv1x1 blocks
  const int FROWS = NB * CH * HH / 2;     // 16384 FFT row blocks
  const int FCOLS = NB * CH * (WW / 16);  // 2048 FFT col blocks

  // 0) weight prep (+ zero bias)
  k_prep<<<64, 256, 0, stream>>>(wq, wk, wv, wd, wc, wi, w1x1b, wcb, wib, zbias);
  // 1) yd = conv1x1(y, wd) + bd
  k_conv1x1<<<C1BLK, 256, 0, stream>>>(y, 32L * HWC, 0, w1x1b + 3 * 1024, 0, bd, yd_att);
  // 2,3) forward FFT with folded fftshift -> dft_pre
  k_fft_rows<<<FROWS, 256, 0, stream>>>(yd_att, tmpc);
  k_fft_cols<<<FCOLS, 256, 0, stream>>>(tmpc, dft_pre);
  // 4) dft = conv3x3(dft_pre, wc) + bc
  k_conv3x3<<<dim3(WW / 64 * HH, NB), 256, 0, stream>>>(dft_pre, wcb, bc, dft, 0);
  // 5) xq / yk / yv 1x1 projections
  k_conv1x1<<<C1BLK, 256, 0, stream>>>(x,   32L * HWC, 0,  w1x1b,            0, bq, xq);
  k_conv1x1<<<C1BLK, 256, 0, stream>>>(dft, 64L * HWC, 0,  w1x1b + 1 * 1024, 0, bk, yk);
  k_conv1x1<<<C1BLK, 256, 0, stream>>>(dft, 64L * HWC, 32, w1x1b + 2 * 1024, 0, bv, yv);
  // 6) attention scores + softmax(axis=c), bf16
  k_att_partial<<<NB * 64, 256, 0, stream>>>(xq, yk, part);
  k_att_reduce<<<NB, 256, 0, stream>>>(part, attbf);
  // 7) attout = att x yv  (zero bias -> branch-free epilogue)
  k_conv1x1<<<C1BLK, 256, 0, stream>>>(yv, 32L * HWC, 0, attbf, 1024, zbias, attout);
  // 8) wiout = conv3x3(relu(dft), wi) + bi
  k_conv3x3<<<dim3(WW / 64 * HH, NB), 256, 0, stream>>>(dft, wib, bi, wiout, 1);
  // 9) inverse FFT (folded ifftshift) + abs + add -> out
  k_ifft_rows<<<FROWS, 256, 0, stream>>>(wiout, tmpc);
  k_ifft_cols<<<FCOLS, 256, 0, stream>>>(tmpc, attout, out);
}
